// TemEnc_38869454029588
// MI455X (gfx1250) — compile-verified, matrix operand order checked by reading, not run on hardware
//
#include <hip/hip_runtime.h>
#include <math.h>

// ---------------- problem constants (match reference) ----------------
#define BB   8
#define TT   1024
#define CIN  55
#define DD   512
#define LL   3
#define SEQW 10
#define TRN  512          // masked count
#define NU   (TT - TRN)   // unmasked count = 512

typedef __attribute__((ext_vector_type(16))) _Float16 v16h;
typedef __attribute__((ext_vector_type(8)))  _Float16 v8h;
typedef __attribute__((ext_vector_type(8)))  float    v8f;

// ---------------- sinusoidal PE on the fly ----------------
__device__ __forceinline__ float pe_val(int t, int d) {
    int i = d >> 1;
    float div = __expf((float)(2 * i) * (-9.210340371976184f / (float)DD)); // ln(10000)
    float ang = (float)t * div;
    return (d & 1) ? __cosf(ang) : __sinf(ang);
}

// ---------------- embedding: circular conv1d(k=3) + PE ----------------
__global__ void embed_kernel(const float* __restrict__ x,
                             const float* __restrict__ w,
                             float* __restrict__ ex) {
    int idx = blockIdx.x * blockDim.x + threadIdx.x;
    if (idx >= BB * TT * DD) return;
    int d = idx % DD;
    int t = (idx / DD) % TT;
    int b = idx / (DD * TT);
    int tm = (t + TT - 1) % TT, tp = (t + 1) % TT;
    const float* xb = x + (size_t)b * TT * CIN;
    const float* wd = w + (size_t)d * CIN * 3;
    float acc = 0.f;
    #pragma unroll 5
    for (int c = 0; c < CIN; ++c) {
        acc += xb[(size_t)tm * CIN + c] * wd[c * 3 + 0]
             + xb[(size_t)t  * CIN + c] * wd[c * 3 + 1]
             + xb[(size_t)tp * CIN + c] * wd[c * 3 + 2];
    }
    ex[idx] = acc + pe_val(t, d);
}

// ---------------- windowed mean/var score, one wave per (b,t) ----------------
__global__ void score_kernel(const float* __restrict__ ex, float* __restrict__ score) {
    int wid  = (blockIdx.x * blockDim.x + threadIdx.x) >> 5;
    int lane = threadIdx.x & 31;
    if (wid >= BB * TT) return;
    int t = wid % TT, b = wid / TT;
    int lo = t + 1 - SEQW; if (lo < 0) lo = 0;
    float inv = 1.f / (float)(t + 1 - lo);
    const float* base = ex + (size_t)b * TT * DD;
    float sm = 0.f, sv = 0.f;
    for (int d = lane; d < DD; d += 32) {
        float s1 = 0.f, s2 = 0.f;
        for (int tau = lo; tau <= t; ++tau) {
            float v = base[(size_t)tau * DD + d];
            s1 += v; s2 += v * v;
        }
        float m = s1 * inv, m2 = s2 * inv;
        sm += m; sv += m2 - m * m;
    }
    #pragma unroll
    for (int off = 16; off; off >>= 1) {
        sm += __shfl_xor(sm, off, 32);
        sv += __shfl_xor(sv, off, 32);
    }
    if (lane == 0) score[wid] = sv / sm;
}

// ---------------- deterministic top-k partition via rank ----------------
__global__ void rank_kernel(const float* __restrict__ score, int* __restrict__ flags) {
    int idx = blockIdx.x * blockDim.x + threadIdx.x;
    if (idx >= BB * TT) return;
    int t = idx % TT, b = idx / TT;
    const float* s = score + (size_t)b * TT;
    float v = s[t];
    int rank = 0;
    for (int j = 0; j < TT; ++j) {
        float sj = s[j];
        rank += (sj > v) || (sj == v && j < t);
    }
    flags[idx] = (rank < TRN) ? 1 : 0;   // 1 = masked
}

__global__ void compact_kernel(const int* __restrict__ flags,
                               int* __restrict__ uslot, int* __restrict__ unmap) {
    int idx = blockIdx.x * blockDim.x + threadIdx.x;
    if (idx >= BB * TT) return;
    int t = idx % TT, b = idx / TT;
    const int* f = flags + (size_t)b * TT;
    int pos = 0;
    for (int j = 0; j < t; ++j) pos += (f[j] == 0);
    if (f[t] == 0) { uslot[idx] = pos; unmap[(size_t)b * NU + pos] = t; }
    else           { uslot[idx] = -1; }
}

// gather unmasked tokens: f32 (residual use) + f16 (GEMM A operand)
__global__ void gather_kernel(const float* __restrict__ ex, const int* __restrict__ unmap,
                              float* __restrict__ xenc, _Float16* __restrict__ xench) {
    int idx = blockIdx.x * blockDim.x + threadIdx.x;
    if (idx >= BB * NU * DD) return;
    int d = idx % DD;
    int slot = (idx / DD) % NU;
    int b = idx / (DD * NU);
    int t = unmap[(size_t)b * NU + slot];
    float v = ex[((size_t)b * TT + t) * DD + d];
    xenc[idx] = v;
    xench[idx] = (_Float16)v;
}

// build decoder tokens (in place over ex) + f16 copy
__global__ void scatter_kernel(float* __restrict__ tokens, _Float16* __restrict__ tokensh,
                               const float* __restrict__ ux,
                               const int* __restrict__ uslot, const float* __restrict__ mtok) {
    int idx = blockIdx.x * blockDim.x + threadIdx.x;
    if (idx >= BB * TT * DD) return;
    int d = idx % DD;
    int t = (idx / DD) % TT;
    int b = idx / (DD * TT);
    int s = uslot[(size_t)b * TT + t];
    float v = (s >= 0) ? ux[((size_t)b * NU + s) * DD + d]
                       : (mtok[d] + pe_val(t, d));
    tokens[idx] = v;
    tokensh[idx] = (_Float16)v;
}

// ---------------- fp32 -> fp16 transpose (for weights: [K,N] -> [N,K]) ----------------
__global__ void cvt_transpose_kernel(const float* __restrict__ in, _Float16* __restrict__ outT,
                                     int Rr, int Cc) {
    int idx = blockIdx.x * blockDim.x + threadIdx.x;
    if (idx >= Rr * Cc) return;
    int c = idx % Cc, r = idx / Cc;
    outT[(size_t)c * Rr + r] = (_Float16)in[idx];
}

// ---------------- softmax: one 256-thread block per row; dual f32 + f16 out ----------------
__global__ void softmax_kernel(const float* __restrict__ S, float* __restrict__ P,
                               _Float16* __restrict__ Ph, int N) {
    int row = blockIdx.x;
    const float* in = S + (size_t)row * N;
    float* out = P + (size_t)row * N;
    _Float16* outh = Ph + (size_t)row * N;
    __shared__ float red[256];
    float mx = -INFINITY;
    for (int i = threadIdx.x; i < N; i += 256) mx = fmaxf(mx, in[i]);
    red[threadIdx.x] = mx; __syncthreads();
    for (int s = 128; s; s >>= 1) {
        if (threadIdx.x < s) red[threadIdx.x] = fmaxf(red[threadIdx.x], red[threadIdx.x + s]);
        __syncthreads();
    }
    mx = red[0]; __syncthreads();
    float sum = 0.f;
    for (int i = threadIdx.x; i < N; i += 256) {
        float e = __expf(in[i] - mx);
        out[i] = e; sum += e;
    }
    red[threadIdx.x] = sum; __syncthreads();
    for (int s = 128; s; s >>= 1) {
        if (threadIdx.x < s) red[threadIdx.x] += red[threadIdx.x + s];
        __syncthreads();
    }
    float inv = 1.f / red[0];
    for (int i = threadIdx.x; i < N; i += 256) {
        float v = out[i] * inv;
        out[i] = v;
        outh[i] = (_Float16)v;
    }
}

// ---------------- layernorm over D=512; dual f32 + f16 out ----------------
__global__ void layernorm_kernel(const float* __restrict__ X, const float* __restrict__ g,
                                 const float* __restrict__ bta,
                                 float* __restrict__ Y, _Float16* __restrict__ Yh) {
    int row = blockIdx.x;
    const float* in = X + (size_t)row * DD;
    float* out = Y + (size_t)row * DD;
    _Float16* outh = Yh + (size_t)row * DD;
    __shared__ float red[128];
    float s = 0.f;
    for (int i = threadIdx.x; i < DD; i += 128) s += in[i];
    red[threadIdx.x] = s; __syncthreads();
    for (int st = 64; st; st >>= 1) { if (threadIdx.x < st) red[threadIdx.x] += red[threadIdx.x + st]; __syncthreads(); }
    float mu = red[0] / (float)DD; __syncthreads();
    float v = 0.f;
    for (int i = threadIdx.x; i < DD; i += 128) { float d = in[i] - mu; v += d * d; }
    red[threadIdx.x] = v; __syncthreads();
    for (int st = 64; st; st >>= 1) { if (threadIdx.x < st) red[threadIdx.x] += red[threadIdx.x + st]; __syncthreads(); }
    float rstd = rsqrtf(red[0] / (float)DD + 1e-5f);
    for (int i = threadIdx.x; i < DD; i += 128) {
        float y = (in[i] - mu) * rstd * g[i] + bta[i];
        out[i] = y;
        outh[i] = (_Float16)y;
    }
}

// ---------------- WMMA fragment load: two contiguous b128 loads per lane ----------------
__device__ __forceinline__ v16h load_frag(const _Float16* __restrict__ p, int half) {
    v8h a0 = *(const v8h*)(p + 8 * half);        // K = 8h .. 8h+7   -> elems 0..7
    v8h a1 = *(const v8h*)(p + 16 + 8 * half);   // K = 16+8h .. +7  -> elems 8..15
    v16h r;
    #pragma unroll
    for (int j = 0; j < 8; ++j) { r[j] = a0[j]; r[j + 8] = a1[j]; }
    return r;
}

// ---------------- WMMA GEMM: C = act(scale * A @ Bt^T + bias + R) ----------------
// A : f16 [M,K] row-major (lda).  Bt : f16 [N,K] row-major (ldb)  (i.e. B pre-transposed).
// One wave computes a 32(M) x 64(N) tile: 8 WMMAs per 32-K step, 12 b128 loads.
// OUTMODE: 0 = f32 (Cf, ldc); 1 = f16 (Ch, ldc); 2 = f16 transposed (Ch[n*ldct+m]);
//          3 = f32 + f16 straight.
// ACT: 0 none, 1 exact-GELU, 2 sigmoid.
template <int ACT, int OUTMODE>
__global__ void gemm_wmma(const _Float16* __restrict__ A, long sA,
                          const _Float16* __restrict__ Bt, long sB,
                          float* __restrict__ Cf, long sC,
                          _Float16* __restrict__ Ch, long sCh,
                          const float* __restrict__ bias,
                          const float* __restrict__ R, long sR,
                          int M, int N, int K,
                          int lda, int ldb, int ldc, int ldct, float scale) {
    int lane = threadIdx.x & 31;
    int rowm = lane & 15;
    int half = lane >> 4;
    int tn0 = blockIdx.x * 64;
    int tm  = blockIdx.y * 32;
    int bz  = blockIdx.z;
    A  += (size_t)bz * sA;
    Bt += (size_t)bz * sB;
    if (Cf) Cf += (size_t)bz * sC;
    if (Ch) Ch += (size_t)bz * sCh;
    if (R)  R  += (size_t)bz * sR;

    v8f acc[2][4] = {};
    const _Float16* arow0 = A + (size_t)(tm + rowm) * lda;
    const _Float16* arow1 = arow0 + (size_t)16 * lda;

    for (int kk = 0; kk < K; kk += 32) {
        v16h af0 = load_frag(arow0 + kk, half);
        v16h af1 = load_frag(arow1 + kk, half);
        #pragma unroll
        for (int u = 0; u < 4; ++u) {
            int col = tn0 + u * 16 + rowm;
            v16h bf = load_frag(Bt + (size_t)col * ldb + kk, half);
            acc[0][u] = __builtin_amdgcn_wmma_f32_16x16x32_f16(
                false, af0, false, bf, (short)0, acc[0][u], false, false);
            acc[1][u] = __builtin_amdgcn_wmma_f32_16x16x32_f16(
                false, af1, false, bf, (short)0, acc[1][u], false, false);
        }
    }

    #pragma unroll
    for (int hm = 0; hm < 2; ++hm) {
        #pragma unroll
        for (int u = 0; u < 4; ++u) {
            int n = tn0 + u * 16 + rowm;
            #pragma unroll
            for (int r = 0; r < 8; ++r) {
                int m = tm + 16 * hm + r + 8 * half;
                float v = acc[hm][u][r] * scale;
                if (bias) v += bias[n];
                if (R)    v += R[(size_t)m * ldc + n];
                if (ACT == 1) v = 0.5f * v * (1.f + erff(v * 0.70710678118654752f));
                else if (ACT == 2) v = 1.f / (1.f + __expf(-v));
                if (OUTMODE == 0 || OUTMODE == 3) Cf[(size_t)m * ldc + n] = v;
                if (OUTMODE == 1 || OUTMODE == 3) Ch[(size_t)m * ldc + n] = (_Float16)v;
                if (OUTMODE == 2)                 Ch[(size_t)n * ldct + m] = (_Float16)v;
            }
        }
    }
}

// ---------------- host orchestration ----------------
extern "C" void kernel_launch(void* const* d_in, const int* in_sizes, int n_in,
                              void* d_out, int out_size, void* d_ws, size_t ws_size,
                              hipStream_t stream) {
    const float* x       = (const float*)d_in[0];
    const float* conv_w  = (const float*)d_in[1];
    const float* mask_tk = (const float*)d_in[2];
    const float* pro_w1  = (const float*)d_in[3];
    const float* pro_b1  = (const float*)d_in[4];
    const float* pro_w2  = (const float*)d_in[5];
    const float* pro_b2  = (const float*)d_in[6];
    const float* eWq = (const float*)d_in[7];
    const float* eWk = (const float*)d_in[8];
    const float* eWv = (const float*)d_in[9];
    const float* eWo = (const float*)d_in[10];
    const float* ebq = (const float*)d_in[11];
    const float* ebk = (const float*)d_in[12];
    const float* ebv = (const float*)d_in[13];
    const float* ebo = (const float*)d_in[14];
    const float* eg  = (const float*)d_in[15];
    const float* eb  = (const float*)d_in[16];
    const float* eng = (const float*)d_in[17];
    const float* enb = (const float*)d_in[18];
    const float* dWq = (const float*)d_in[19];
    const float* dWk = (const float*)d_in[20];
    const float* dWv = (const float*)d_in[21];
    const float* dWo = (const float*)d_in[22];
    const float* dbq = (const float*)d_in[23];
    const float* dbk = (const float*)d_in[24];
    const float* dbv = (const float*)d_in[25];
    const float* dbo = (const float*)d_in[26];
    const float* dg  = (const float*)d_in[27];
    const float* db  = (const float*)d_in[28];
    const float* dng = (const float*)d_in[29];
    const float* dnb = (const float*)d_in[30];

    float* out = (float*)d_out;
    const size_t BTD = (size_t)BB * TT * DD;   // 4M
    const size_t BND = (size_t)BB * NU * DD;   // 2M
    const size_t BTT = (size_t)BB * TT * TT;   // 8M
    const size_t WSZ = (size_t)DD * DD;        // 256K per weight matrix

    // ---- workspace layout: f32 region, int region, f16 region ----
    float* ex    = (float*)d_ws;          // B*T*D (tokens f32)
    float* Ob    = ex + BTD;              // B*T*D
    float* LN    = Ob + BTD;              // B*T*D
    float* Sb    = LN + BTD;              // B*T*T (pre-softmax scores)
    float* Xenc  = Sb + BTT;              // B*NU*D
    float* score = Xenc + BND;            // B*T
    int*   flags = (int*)(score + (size_t)BB * TT);
    int*   uslot = flags + (size_t)BB * TT;
    int*   unmap = uslot + (size_t)BB * TT;
    uintptr_t hraw = (uintptr_t)(unmap + (size_t)BB * NU);
    _Float16* h16 = (_Float16*)((hraw + 63) & ~(uintptr_t)63);
    _Float16* exh   = h16;                // B*T*D  (tokens f16)
    _Float16* Xench = exh + BTD;          // B*NU*D
    _Float16* Qh    = Xench + BND;        // B*T*D  (also proj-head hidden)
    _Float16* Kh    = Qh + BTD;           // B*T*D
    _Float16* Vth   = Kh + BTD;           // [D][B*Ntok] f16 (V transposed)
    _Float16* LNh   = Vth + BTD;          // B*T*D
    _Float16* Ph    = LNh + BTD;          // B*T*T (probs f16)
    _Float16* Wt    = Ph + BTT;           // 26 * D*D (all weights, [N,K] f16)
    // Wt index: enc q:0+l k:3+l v:6+l o:9+l ; dec +12 ; w1:24 w2:25

    const float qk_scale = 0.044194173824159216f;  // 1/sqrt(512)
    const int CG = (int)((WSZ + 255) / 256);

    // ---- one-time (per launch) weight fp32->f16 transposes ----
    for (int l = 0; l < LL; ++l) {
        cvt_transpose_kernel<<<CG, 256, 0, stream>>>(eWq + l * WSZ, Wt + (0 + l) * WSZ, DD, DD);
        cvt_transpose_kernel<<<CG, 256, 0, stream>>>(eWk + l * WSZ, Wt + (3 + l) * WSZ, DD, DD);
        cvt_transpose_kernel<<<CG, 256, 0, stream>>>(eWv + l * WSZ, Wt + (6 + l) * WSZ, DD, DD);
        cvt_transpose_kernel<<<CG, 256, 0, stream>>>(eWo + l * WSZ, Wt + (9 + l) * WSZ, DD, DD);
        cvt_transpose_kernel<<<CG, 256, 0, stream>>>(dWq + l * WSZ, Wt + (12 + l) * WSZ, DD, DD);
        cvt_transpose_kernel<<<CG, 256, 0, stream>>>(dWk + l * WSZ, Wt + (15 + l) * WSZ, DD, DD);
        cvt_transpose_kernel<<<CG, 256, 0, stream>>>(dWv + l * WSZ, Wt + (18 + l) * WSZ, DD, DD);
        cvt_transpose_kernel<<<CG, 256, 0, stream>>>(dWo + l * WSZ, Wt + (21 + l) * WSZ, DD, DD);
    }
    cvt_transpose_kernel<<<CG, 256, 0, stream>>>(pro_w1, Wt + 24 * WSZ, DD, DD);
    cvt_transpose_kernel<<<CG, 256, 0, stream>>>(pro_w2, Wt + 25 * WSZ, DD, DD);

    // 1) embedding, 2) score, 3) partition + gather
    embed_kernel<<<(BB * TT * DD + 255) / 256, 256, 0, stream>>>(x, conv_w, ex);
    score_kernel<<<(BB * TT * 32 + 255) / 256, 256, 0, stream>>>(ex, score);
    rank_kernel<<<(BB * TT + 255) / 256, 256, 0, stream>>>(score, flags);
    compact_kernel<<<(BB * TT + 255) / 256, 256, 0, stream>>>(flags, uslot, unmap);
    gather_kernel<<<(BB * NU * DD + 255) / 256, 256, 0, stream>>>(ex, unmap, Xenc, Xench);

    // ---------------- encoder on unmasked (N = NU = 512) ----------------
    {
        const int Nt = NU;
        const long sQ = (long)Nt * DD, sS = (long)Nt * Nt;
        for (int l = 0; l < LL; ++l) {
            const _Float16* WtQ = Wt + (0 + l) * WSZ;
            const _Float16* WtK = Wt + (3 + l) * WSZ;
            const _Float16* WtV = Wt + (6 + l) * WSZ;
            const _Float16* WtO = Wt + (9 + l) * WSZ;
            dim3 gP(DD / 64, (BB * Nt) / 32, 1);
            gemm_wmma<0, 1><<<gP, 32, 0, stream>>>(Xench, 0, WtQ, 0, (float*)nullptr, 0, Qh, 0,
                ebq + l * DD, nullptr, 0, BB * Nt, DD, DD, DD, DD, DD, 0, 1.f);
            gemm_wmma<0, 1><<<gP, 32, 0, stream>>>(Xench, 0, WtK, 0, (float*)nullptr, 0, Kh, 0,
                ebk + l * DD, nullptr, 0, BB * Nt, DD, DD, DD, DD, DD, 0, 1.f);
            gemm_wmma<0, 2><<<gP, 32, 0, stream>>>(Xench, 0, WtV, 0, (float*)nullptr, 0, Vth, 0,
                ebv + l * DD, nullptr, 0, BB * Nt, DD, DD, DD, DD, DD, BB * Nt, 1.f);
            dim3 gS(Nt / 64, Nt / 32, BB);
            gemm_wmma<0, 0><<<gS, 32, 0, stream>>>(Qh, sQ, Kh, sQ, Sb, sS, (_Float16*)nullptr, 0,
                nullptr, nullptr, 0, Nt, Nt, DD, DD, DD, Nt, 0, qk_scale);
            softmax_kernel<<<BB * Nt, 256, 0, stream>>>(Sb, Sb, Ph, Nt);
            dim3 gO(DD / 64, Nt / 32, BB);
            gemm_wmma<0, 0><<<gO, 32, 0, stream>>>(Ph, sS, Vth, (long)Nt, Ob, sQ, (_Float16*)nullptr, 0,
                nullptr, Xenc, sQ, Nt, DD, Nt, Nt, BB * Nt, DD, 0, 1.f);
            layernorm_kernel<<<BB * Nt, 128, 0, stream>>>(Ob, eg + l * DD, eb + l * DD, LN, LNh);
            gemm_wmma<0, 3><<<gP, 32, 0, stream>>>(LNh, 0, WtO, 0, Xenc, 0, Xench, 0,
                ebo + l * DD, Ob, 0, BB * Nt, DD, DD, DD, DD, DD, 0, 1.f);
        }
        layernorm_kernel<<<BB * Nt, 128, 0, stream>>>(Xenc, eng, enb, LN, LNh);  // ux -> LN (f32)
    }

    // 4) scatter: decoder tokens, f32 + f16
    scatter_kernel<<<(BB * TT * DD + 255) / 256, 256, 0, stream>>>(ex, exh, LN, uslot, mask_tk);

    // ---------------- decoder on full sequence (N = T = 1024) ----------------
    {
        const int Nt = TT;
        const long sQ = (long)Nt * DD, sS = (long)Nt * Nt;
        for (int l = 0; l < LL; ++l) {
            const _Float16* WtQ = Wt + (12 + l) * WSZ;
            const _Float16* WtK = Wt + (15 + l) * WSZ;
            const _Float16* WtV = Wt + (18 + l) * WSZ;
            const _Float16* WtO = Wt + (21 + l) * WSZ;
            float* attn = out + (size_t)l * BTT;
            dim3 gP(DD / 64, (BB * Nt) / 32, 1);
            gemm_wmma<0, 1><<<gP, 32, 0, stream>>>(exh, 0, WtQ, 0, (float*)nullptr, 0, Qh, 0,
                dbq + l * DD, nullptr, 0, BB * Nt, DD, DD, DD, DD, DD, 0, 1.f);
            gemm_wmma<0, 1><<<gP, 32, 0, stream>>>(exh, 0, WtK, 0, (float*)nullptr, 0, Kh, 0,
                dbk + l * DD, nullptr, 0, BB * Nt, DD, DD, DD, DD, DD, 0, 1.f);
            gemm_wmma<0, 2><<<gP, 32, 0, stream>>>(exh, 0, WtV, 0, (float*)nullptr, 0, Vth, 0,
                dbv + l * DD, nullptr, 0, BB * Nt, DD, DD, DD, DD, DD, BB * Nt, 1.f);
            dim3 gS(Nt / 64, Nt / 32, BB);
            gemm_wmma<0, 0><<<gS, 32, 0, stream>>>(Qh, sQ, Kh, sQ, Sb, sS, (_Float16*)nullptr, 0,
                nullptr, nullptr, 0, Nt, Nt, DD, DD, DD, Nt, 0, qk_scale);
            softmax_kernel<<<BB * Nt, 256, 0, stream>>>(Sb, attn, Ph, Nt);  // probs -> d_out slice
            dim3 gO(DD / 64, Nt / 32, BB);
            gemm_wmma<0, 0><<<gO, 32, 0, stream>>>(Ph, sS, Vth, (long)Nt, Ob, sQ, (_Float16*)nullptr, 0,
                nullptr, ex, sQ, Nt, DD, Nt, Nt, BB * Nt, DD, 0, 1.f);
            layernorm_kernel<<<BB * Nt, 128, 0, stream>>>(Ob, dg + l * DD, db + l * DD, LN, LNh);
            gemm_wmma<0, 3><<<gP, 32, 0, stream>>>(LNh, 0, WtO, 0, ex, 0, exh, 0,
                dbo + l * DD, Ob, 0, BB * Nt, DD, DD, DD, DD, DD, 0, 1.f);
        }
        layernorm_kernel<<<BB * Nt, 128, 0, stream>>>(ex, dng, dnb, LN, LNh);  // dx

        // projection head: H = gelu(LN@w1+b1) -> Qh (f16), rec = sigmoid(H@w2+b2) -> out
        dim3 gP(DD / 64, (BB * Nt) / 32, 1);
        gemm_wmma<1, 1><<<gP, 32, 0, stream>>>(LNh, 0, Wt + 24 * WSZ, 0, (float*)nullptr, 0, Qh, 0,
            pro_b1, nullptr, 0, BB * Nt, DD, DD, DD, DD, DD, 0, 1.f);
        float* rec = out + 3 * BTT;
        gemm_wmma<2, 0><<<gP, 32, 0, stream>>>(Qh, 0, Wt + 25 * WSZ, 0, rec, 0, (_Float16*)nullptr, 0,
            pro_b2, nullptr, 0, BB * Nt, DD, DD, DD, DD, DD, 0, 1.f);
    }
}
